// DenseEdgeAggregator_46033459479182
// MI455X (gfx1250) — compile-verified, matrix operand order checked by reading, not run on hardware
//
#include <hip/hip_runtime.h>
#include <hip/hip_bf16.h>
#include <stdint.h>

// Problem constants (match reference)
#define N_NODES 1024
#define IN_DIM  256
#define EDGE_DIM 32
#define HID      32
#define OUT_DIM 128

typedef __bf16 bf16_t;
typedef __attribute__((ext_vector_type(16))) bf16_t v16bf;
typedef __attribute__((ext_vector_type(8)))  float  v8f;
typedef __attribute__((ext_vector_type(4)))  unsigned int v4u;
typedef __attribute__((ext_vector_type(8)))  int v8i;
typedef __attribute__((ext_vector_type(4)))  int v4i;

// -------------------- small device helpers --------------------

__device__ __forceinline__ bf16_t f2bf(float f) {
  // native hardware convert (gfx1250 has bf16 cvt); clang lowers fptrunc->bf16
  return (bf16_t)f;
}

__device__ __forceinline__ void lds_wave_sync() {
  // wave-internal LDS producer->consumer ordering (writes by some lanes,
  // reads by others, same wave). CDNA5 split counter: DScnt.
  asm volatile("s_wait_dscnt 0x0" ::: "memory");
}

// ---- Tensor Data Mover: 1-D copy of 512 f32 (2KB) global -> LDS ----
// D# per CDNA5 ISA ch.8: group0 = {count/flags, lds_addr, global_addr, type=2},
// group1 = {data_size=4B, tensor_dim0=512, tile_dim0=512, stride0=512}.
// LDS byte address taken as low 32 bits of the flat pointer (LDS aperture
// truncation per ISA 10.2). This toolchain exposes the 6-arg builtin
// (clang-23 form): (v4u, v8i, v4i, v4i, v8i, i32 cpol); trailing group unused
// for <=2-D tensors.
__device__ __forceinline__ void tdm_copy_512f(float* lds_dst, const float* gsrc) {
#if __has_builtin(__builtin_amdgcn_tensor_load_to_lds)
  uint64_t ga = (uint64_t)(uintptr_t)gsrc;
  uint32_t la = (uint32_t)(uintptr_t)lds_dst;
  v4u g0;
  g0[0] = 1u;                                   // count=1, user descriptor
  g0[1] = la;                                   // lds_addr (bytes)
  g0[2] = (uint32_t)ga;                         // global_addr[31:0]
  g0[3] = (uint32_t)((ga >> 32) & 0x01FFFFFFu)  // global_addr[56:32]
          | 0x80000000u;                        // type=2 ("image")
  v8i g1;
  g1[0] = (int)(2u << 16);      // workgroup_mask=0, data_size=2 (4 bytes)
  g1[1] = (int)(512u << 16);    // tensor_dim0[15:0]=512 (atomic_barrier_addr=0)
  g1[2] = (int)(1u << 16);      // tensor_dim0[31:16]=0, tensor_dim1[15:0]=1
  g1[3] = (int)(512u << 16);    // tensor_dim1[31:16]=0, tile_dim0=512
  g1[4] = 0;                    // tile_dim1=0 (1-D), tile_dim2=0
  g1[5] = 512;                  // tensor_dim0_stride[31:0]
  g1[6] = 0;                    // stride0[47:32]=0, stride1[15:0]=0
  g1[7] = 0;
  v4i gz4; gz4[0] = gz4[1] = gz4[2] = gz4[3] = 0;   // groups 2/3 unused (<=2D)
  v8i gz8;
#pragma unroll
  for (int i = 0; i < 8; ++i) gz8[i] = 0;           // trailing group unused
  __builtin_amdgcn_tensor_load_to_lds(g0, g1, gz4, gz4, gz8, 0);
#else
  int lane = threadIdx.x & 31;
  for (int t = lane; t < 512; t += 32) lds_dst[t] = gsrc[t];
  lds_wave_sync();
#endif
}

__device__ __forceinline__ void tdm_wait_le1() {
#if __has_builtin(__builtin_amdgcn_tensor_load_to_lds)
  __builtin_amdgcn_s_wait_tensorcnt(1);
#endif
  asm volatile("" ::: "memory");
}
__device__ __forceinline__ void tdm_wait0() {
#if __has_builtin(__builtin_amdgcn_tensor_load_to_lds)
  __builtin_amdgcn_s_wait_tensorcnt(0);
#endif
  asm volatile("" ::: "memory");
}

// K-offset within a 32-wide K window for bf16 A/B operand register pair j
// (per CDNA5 ISA 16-bit A-matrix 16x32 layout; B assumed symmetric over N).
__device__ __forceinline__ int kbase(int j, int half) {
  return ((j & 3) * 2) + ((j >> 2) * 16) + half * 8;
}

__device__ __forceinline__ v8f wmma_bf16(v16bf a, v16bf b, v8f c) {
  return __builtin_amdgcn_wmma_f32_16x16x32_bf16(
      /*neg_a=*/false, a, /*neg_b=*/false, b,
      /*c_mod=*/(short)0, c, /*reuse_a=*/false, /*reuse_b=*/false);
}

// A tile (16 rows x 32 K) from row-major f32 source, converted to bf16.
__device__ __forceinline__ v16bf load_A_f32(const float* src, int ld, int row0, int k0) {
  int l = threadIdx.x & 31, m = l & 15, half = l >> 4;
  const float* p = src + (size_t)(row0 + m) * ld + k0;
  v16bf a;
#pragma unroll
  for (int j = 0; j < 8; ++j) {
    int kb = kbase(j, half);
    a[2 * j]     = f2bf(p[kb]);
    a[2 * j + 1] = f2bf(p[kb + 1]);
  }
  return a;
}

// B tile (32 K x 16 cols) from row-major f32 weight matrix.
__device__ __forceinline__ v16bf load_B_f32(const float* W, int ld, int k0, int o0) {
  int l = threadIdx.x & 31, n = l & 15, half = l >> 4;
  v16bf b;
#pragma unroll
  for (int j = 0; j < 8; ++j) {
    int kb = kbase(j, half);
    b[2 * j]     = f2bf(W[(size_t)(k0 + kb) * ld + o0 + n]);
    b[2 * j + 1] = f2bf(W[(size_t)(k0 + kb + 1) * ld + o0 + n]);
  }
  return b;
}

// B tile from a pre-converted bf16 row-major matrix.
__device__ __forceinline__ v16bf load_B_bf16(const bf16_t* W, int ld, int k0, int o0) {
  int l = threadIdx.x & 31, n = l & 15, half = l >> 4;
  v16bf b;
#pragma unroll
  for (int j = 0; j < 8; ++j) {
    int kb = kbase(j, half);
    b[2 * j]     = W[(size_t)(k0 + kb) * ld + o0 + n];
    b[2 * j + 1] = W[(size_t)(k0 + kb + 1) * ld + o0 + n];
  }
  return b;
}

// C initialized with a per-column bias (C layout: col = lane%16).
__device__ __forceinline__ v8f bias_C(const float* bias) {
  float v = bias[threadIdx.x & 15];
  v8f c;
#pragma unroll
  for (int i = 0; i < 8; ++i) c[i] = v;
  return c;
}

// Store C tile (16x16 f32) into row-major f32 destination.
__device__ __forceinline__ void store_C_f32(float* dst, int ld, int row0, int o0, v8f c) {
  int l = threadIdx.x & 31, n = l & 15, half = l >> 4;
#pragma unroll
  for (int i = 0; i < 8; ++i)
    dst[(size_t)(row0 + i + 8 * half) * ld + o0 + n] = c[i];
}

__device__ __forceinline__ void store_C_bf16(bf16_t* dst, int ld, int row0, int o0, v8f c) {
  int l = threadIdx.x & 31, n = l & 15, half = l >> 4;
#pragma unroll
  for (int i = 0; i < 8; ++i)
    dst[(size_t)(row0 + i + 8 * half) * ld + o0 + n] = f2bf(c[i]);
}

// -------------------- kernel A: dense precompute --------------------
// grid = 64 (16-row tiles), block = 192 (6 waves):
//  wave0: x_att = tanh(x@Wx1+bx1)@Wx2+bx2         -> f32 scratch
//  wave1: neib_att (same with Wn*)                -> f32 scratch
//  wave2/3: value = neigh@Wv+bv                   -> bf16 scratch (WMMA B reuse)
//  wave4/5: out[:, :128] = x@Wfx+bfx              -> d_out directly

__global__ __launch_bounds__(192) void precompute_kernel(
    const float* __restrict__ x, const float* __restrict__ neigh,
    const float* __restrict__ Wx1, const float* __restrict__ bx1,
    const float* __restrict__ Wx2, const float* __restrict__ bx2,
    const float* __restrict__ Wn1, const float* __restrict__ bn1,
    const float* __restrict__ Wn2, const float* __restrict__ bn2,
    const float* __restrict__ Wv,  const float* __restrict__ bv,
    const float* __restrict__ Wfx, const float* __restrict__ bfx,
    float* __restrict__ xatt_s, float* __restrict__ natt_s,
    bf16_t* __restrict__ value_bf, float* __restrict__ out) {
  __shared__ float bounce[2][16 * 33];  // C-layout -> A-layout bounce for the 2 MLPs
  const int row0 = blockIdx.x * 16;
  const int w = threadIdx.x >> 5;

  if (w < 2) {
    const float* in = (w == 0) ? x : neigh;
    const float* W1 = (w == 0) ? Wx1 : Wn1;
    const float* b1 = (w == 0) ? bx1 : bn1;
    const float* W2 = (w == 0) ? Wx2 : Wn2;
    const float* b2 = (w == 0) ? bx2 : bn2;
    float* dst = (w == 0) ? xatt_s : natt_s;
    float* B = bounce[w];
    // stage 1: h = tanh(in_tile @ W1 + b1)   (16 x 32)
#pragma unroll
    for (int t = 0; t < 2; ++t) {
      v8f c = bias_C(b1 + t * 16);
      for (int ks = 0; ks < 8; ++ks)
        c = wmma_bf16(load_A_f32(in, IN_DIM, row0, ks * 32),
                      load_B_f32(W1, HID, ks * 32, t * 16), c);
#pragma unroll
      for (int i = 0; i < 8; ++i) c[i] = tanhf(c[i]);
      store_C_f32(B, 33, 0, t * 16, c);
    }
    lds_wave_sync();
    // stage 2: att = h @ W2 + b2             (16 x 32)
#pragma unroll
    for (int t = 0; t < 2; ++t) {
      v8f c = bias_C(b2 + t * 16);
      c = wmma_bf16(load_A_f32(B, 33, 0, 0), load_B_f32(W2, HID, 0, t * 16), c);
      store_C_f32(dst, HID, row0, t * 16, c);
    }
  } else if (w < 4) {
    const int obase = (w - 2) * 64;
#pragma unroll
    for (int t = 0; t < 4; ++t) {
      const int o0 = obase + t * 16;
      v8f c = bias_C(bv + o0);
      for (int ks = 0; ks < 8; ++ks)
        c = wmma_bf16(load_A_f32(neigh, IN_DIM, row0, ks * 32),
                      load_B_f32(Wv, OUT_DIM, ks * 32, o0), c);
      store_C_bf16(value_bf, OUT_DIM, row0, o0, c);
    }
  } else if (w < 6) {
    const int obase = (w - 4) * 64;
#pragma unroll
    for (int t = 0; t < 4; ++t) {
      const int o0 = obase + t * 16;
      v8f c = bias_C(bfx + o0);
      for (int ks = 0; ks < 8; ++ks)
        c = wmma_bf16(load_A_f32(x, IN_DIM, row0, ks * 32),
                      load_B_f32(Wfx, OUT_DIM, ks * 32, o0), c);
      store_C_f32(out, 2 * OUT_DIM, row0, o0, c);  // columns [0,128)
    }
  }
}

// -------------------- kernel B: attention + aggregation --------------------
// grid = 64 (16 rows per WG), block = 256 (8 waves, 2 rows per wave).
// Dynamic LDS layout (floats):
//   S[16][1024]     raw scores, then unnormalized exp weights
//   xatt[16][32]    this tile's x_att
//   rowsum[16]
//   accE[16][32]    sum_k p[k]*edges[k][:]  (unnormalized)
//   E[8][16*33]     per-wave tanh(edges@We) bounce tile
//   NB[8][2][16*32] per-wave double-buffered TDM-staged neib_att tiles

#define SM_S     0
#define SM_XATT  (16 * 1024)
#define SM_RSUM  (SM_XATT + 16 * 32)
#define SM_ACCE  (SM_RSUM + 16)
#define SM_EW    (SM_ACCE + 16 * 32)
#define SM_NB    (SM_EW + 8 * 16 * 33)
#define SM_TOTAL (SM_NB + 8 * 2 * 16 * 32)

__global__ __launch_bounds__(256) void attn_kernel(
    const float* __restrict__ edge_emb, const int* __restrict__ adj,
    const float* __restrict__ We, const float* __restrict__ be,
    const float* __restrict__ Wfe, const float* __restrict__ bfe,
    const float* __restrict__ xatt_s, const float* __restrict__ natt_s,
    const bf16_t* __restrict__ value_bf, float* __restrict__ out) {
  extern __shared__ float sm[];
  float* S     = sm + SM_S;
  float* xatt  = sm + SM_XATT;
  float* rsum  = sm + SM_RSUM;
  float* accE  = sm + SM_ACCE;

  const int base = blockIdx.x * 16;
  const int w = threadIdx.x >> 5;
  const int lane = threadIdx.x & 31;
  const int m = lane & 15, half = lane >> 4;

  // stage tile's x_att into LDS
  for (int t = threadIdx.x; t < 16 * HID; t += 256)
    xatt[t] = xatt_s[(size_t)base * HID + t];
  __syncthreads();

  float* E  = sm + SM_EW + w * (16 * 33);
  float* nb = sm + SM_NB + w * (2 * 16 * HID);  // double buffer, 512 f32 each
  const v16bf BWe0 = load_B_f32(We, HID, 0, 0);
  const v16bf BWe1 = load_B_f32(We, HID, 0, 16);
  const float be0 = be[m], be1 = be[m + 16];

  for (int r = 0; r < 2; ++r) {
    const int nl = 2 * w + r;       // local row
    const int n  = base + nl;       // global node
    const int* arow = adj + (size_t)n * N_NODES;
    const float* xa = xatt + nl * HID;
    float mx = -3.4e38f;

    // prologue: TDM-stage neib_att tile 0 (rows 0..15, 2KB contiguous)
    tdm_copy_512f(nb, natt_s);

    // ---- phase 1: raw scores over all k, 16-k WMMA tiles ----
    for (int kt = 0; kt < 64; ++kt) {
      // kick off async TDM for the next neib_att tile (overlaps gather+WMMA)
      if (kt + 1 < 64)
        tdm_copy_512f(nb + ((kt + 1) & 1) * 512, natt_s + (size_t)(kt + 1) * 512);

      const int kk = kt * 16 + m;
      const int idx = arow[kk];
      const float* ep = edge_emb + (size_t)idx * EDGE_DIM;
      // prefetch next tile's random edge rows into cache (global_prefetch_b8)
      if (kt + 1 < 64)
        __builtin_prefetch(edge_emb + (size_t)arow[kk + 16] * EDGE_DIM, 0, 0);

      v16bf a;
#pragma unroll
      for (int j = 0; j < 8; ++j) {
        int kb = kbase(j, half);
        float2 f = *(const float2*)(ep + kb);
        a[2 * j]     = f2bf(f.x);
        a[2 * j + 1] = f2bf(f.y);
      }
      v8f c0, c1;
#pragma unroll
      for (int i = 0; i < 8; ++i) { c0[i] = be0; c1[i] = be1; }
      c0 = wmma_bf16(a, BWe0, c0);
      c1 = wmma_bf16(a, BWe1, c1);
#pragma unroll
      for (int i = 0; i < 8; ++i) { c0[i] = tanhf(c0[i]); c1[i] = tanhf(c1[i]); }
      // C-layout -> padded LDS tile (rows = k within tile, cols = hid)
#pragma unroll
      for (int i = 0; i < 8; ++i) {
        E[(i + 8 * half) * 33 + m]      = c0[i];
        E[(i + 8 * half) * 33 + m + 16] = c1[i];
      }
      lds_wave_sync();
      // make sure TDM tile kt has landed (<=1 outstanding keeps kt+1 in flight)
      if (kt + 1 < 64) tdm_wait_le1(); else tdm_wait0();

      // fold: score = sum_h (e_att[k][h] + neib_att[k][h]) * x_att[n][h]
      const float* na = nb + (kt & 1) * 512 + m * HID;
      float s = 0.f;
#pragma unroll
      for (int h = half * 16; h < half * 16 + 16; ++h)
        s += (E[m * 33 + h] + na[h]) * xa[h];
      s += __shfl_xor(s, 16);
      s = (s > 0.f) ? s : 0.01f * s;       // leaky_relu(0.01)
      if (idx <= 0) s = -9.0e15f;          // mask adj==0
      mx = fmaxf(mx, s);
      if (half == 0) S[nl * N_NODES + kk] = s;
    }
#pragma unroll
    for (int off = 1; off < 32; off <<= 1) mx = fmaxf(mx, __shfl_xor(mx, off));
    lds_wave_sync();

    // ---- phase 2a: unnormalized softmax weights + row sum ----
    float sum = 0.f;
    for (int k = lane; k < N_NODES; k += 32) {
      float p = __expf(S[nl * N_NODES + k] - mx);
      S[nl * N_NODES + k] = p;
      sum += p;
    }
#pragma unroll
    for (int off = 1; off < 32; off <<= 1) sum += __shfl_xor(sum, off);
    if (lane == 0) rsum[nl] = sum;
    lds_wave_sync();

    // accE[d] = sum_k p[k] * edge_emb[adj[n,k]][d]; lane owns dim d (coalesced gather)
    float acc = 0.f;
    for (int k = 0; k < N_NODES; ++k) {
      float p = S[nl * N_NODES + k];
      int idx = arow[k];
      acc += p * edge_emb[(size_t)idx * EDGE_DIM + lane];
    }
    accE[nl * EDGE_DIM + lane] = acc;
  }
  __syncthreads();

  // ---- phase 2b: agg = (P@value + accE@Wfe)/rowsum + bfe; wave w -> o-tile w ----
  v8f cV;
#pragma unroll
  for (int i = 0; i < 8; ++i) cV[i] = 0.f;
  for (int kt = 0; kt < 32; ++kt)
    cV = wmma_bf16(load_A_f32(S, N_NODES, 0, kt * 32),
                   load_B_bf16(value_bf, OUT_DIM, kt * 32, w * 16), cV);
  v8f cE;
#pragma unroll
  for (int i = 0; i < 8; ++i) cE[i] = 0.f;
  cE = wmma_bf16(load_A_f32(accE, EDGE_DIM, 0, 0),
                 load_B_f32(Wfe, OUT_DIM, 0, w * 16), cE);

  const float bo = bfe[w * 16 + m];
#pragma unroll
  for (int i = 0; i < 8; ++i) {
    const int rr = i + 8 * half;
    float res = (cV[i] + cE[i]) / rsum[rr] + bo;
    out[(size_t)(base + rr) * (2 * OUT_DIM) + OUT_DIM + w * 16 + m] = res;
  }
}

// -------------------- host launch --------------------

extern "C" void kernel_launch(void* const* d_in, const int* in_sizes, int n_in,
                              void* d_out, int out_size, void* d_ws, size_t ws_size,
                              hipStream_t stream) {
  (void)in_sizes; (void)n_in; (void)out_size; (void)ws_size;
  const float* x        = (const float*)d_in[0];
  const float* neigh    = (const float*)d_in[1];
  const float* edge_emb = (const float*)d_in[2];
  const int*   adj      = (const int*)d_in[3];
  const float* Wx1 = (const float*)d_in[4];  const float* bx1 = (const float*)d_in[5];
  const float* Wx2 = (const float*)d_in[6];  const float* bx2 = (const float*)d_in[7];
  const float* Wn1 = (const float*)d_in[8];  const float* bn1 = (const float*)d_in[9];
  const float* Wn2 = (const float*)d_in[10]; const float* bn2 = (const float*)d_in[11];
  const float* We  = (const float*)d_in[12]; const float* be  = (const float*)d_in[13];
  const float* Wv  = (const float*)d_in[14]; const float* bv  = (const float*)d_in[15];
  const float* Wfe = (const float*)d_in[16]; const float* bfe = (const float*)d_in[17];
  const float* Wfx = (const float*)d_in[18]; const float* bfx = (const float*)d_in[19];

  float* out = (float*)d_out;

  // scratch: x_att (1024x32 f32), neib_att (1024x32 f32), value (1024x128 bf16)
  float* xatt_s = (float*)d_ws;
  float* natt_s = xatt_s + N_NODES * HID;
  bf16_t* value_bf = (bf16_t*)(natt_s + N_NODES * HID);

  precompute_kernel<<<N_NODES / 16, 192, 0, stream>>>(
      x, neigh, Wx1, bx1, Wx2, bx2, Wn1, bn1, Wn2, bn2, Wv, bv, Wfx, bfx,
      xatt_s, natt_s, value_bf, out);

  const size_t smem = SM_TOTAL * sizeof(float);  // ~117 KB, fits 320 KB WGP LDS
  attn_kernel<<<N_NODES / 16, 256, smem, stream>>>(
      edge_emb, adj, We, be, Wfe, bfe, xatt_s, natt_s, value_bf, out);
}
